// BinaryLinear_78889959293073
// MI455X (gfx1250) — matrix operand order repaired and checked by reading, not compiled, not run on hardware
//
#include <hip/hip_runtime.h>
#include <stdint.h>

#define BATCH 8192
#define DIN   4096
#define DOUT  4096

#define BM 128
#define BN 256
#define BK 32
#define LDK 40   // padded LDS row (elements): 80 bytes -> conflict-free, 16B aligned

typedef __attribute__((ext_vector_type(16))) __bf16 v16bf;
typedef __attribute__((ext_vector_type(8)))  float  v8f;
typedef int v4i __attribute__((__vector_size__(16)));   // matches builtin param type

#define AS_GLOBAL __attribute__((address_space(1)))
#define AS_LDS    __attribute__((address_space(3)))

struct Frag32 { uint4 lo, hi; };   // 32 bytes == v16bf

#if defined(__has_builtin)
#if __has_builtin(__builtin_amdgcn_global_load_async_to_lds_b128)
#define HAVE_ASYNC_LDS 1
#endif
#if __has_builtin(__builtin_amdgcn_s_wait_asynccnt)
#define HAVE_WAIT_ASYNC 1
#endif
#endif

__device__ __forceinline__ void wait_async0() {
#if defined(HAVE_WAIT_ASYNC)
  __builtin_amdgcn_s_wait_asynccnt(0);
#elif defined(HAVE_ASYNC_LDS)
  asm volatile("s_wait_asynccnt 0" ::: "memory");
#endif
}

__device__ __forceinline__ float sgnf(float t) {
  return (t > 0.f) ? 1.f : ((t < 0.f) ? -1.f : 0.f);
}

// ---------------- x : f32 -> bf16 (round to nearest even) ----------------
__global__ void cvt_x_bf16(const float* __restrict__ x, unsigned short* __restrict__ xb) {
  size_t i = ((size_t)blockIdx.x * blockDim.x + threadIdx.x) * 8;
  uint4 u0 = *(const uint4*)(x + i);
  uint4 u1 = *(const uint4*)(x + i + 4);
  unsigned int w[8] = {u0.x,u0.y,u0.z,u0.w,u1.x,u1.y,u1.z,u1.w};
  unsigned int r[8];
#pragma unroll
  for (int k = 0; k < 8; ++k) {
    unsigned int u = w[k];
    u += 0x7fffu + ((u >> 16) & 1u);
    r[k] = u >> 16;
  }
  uint4 o;
  o.x = r[0] | (r[1] << 16); o.y = r[2] | (r[3] << 16);
  o.z = r[4] | (r[5] << 16); o.w = r[6] | (r[7] << 16);
  *(uint4*)(xb + i) = o;
}

// ---------------- W : sign(W) -> bf16 {-1, 0, +1} ----------------
__global__ void cvt_w_sign_bf16(const float* __restrict__ W, unsigned short* __restrict__ wb) {
  size_t i = ((size_t)blockIdx.x * blockDim.x + threadIdx.x) * 8;
  float4 f0 = *(const float4*)(W + i);
  float4 f1 = *(const float4*)(W + i + 4);
  float f[8] = {f0.x,f0.y,f0.z,f0.w,f1.x,f1.y,f1.z,f1.w};
  unsigned int r[8];
#pragma unroll
  for (int k = 0; k < 8; ++k)
    r[k] = (f[k] > 0.f) ? 0x3f80u : ((f[k] < 0.f) ? 0xbf80u : 0u);
  uint4 o;
  o.x = r[0] | (r[1] << 16); o.y = r[2] | (r[3] << 16);
  o.z = r[4] | (r[5] << 16); o.w = r[6] | (r[7] << 16);
  *(uint4*)(wb + i) = o;
}

// ---------------- y = xb @ wb^T + bias, bf16 WMMA ----------------
// Block tile 128x256, BK=32. 8 waves: waveM in {0,1} (64 rows), waveN in {0..3} (64 cols).
// Each wave: 4x4 accumulator tiles of 16x16 -> 16 v_wmma per K-step.
__global__ __launch_bounds__(256) void bingemm(
    const unsigned short* __restrict__ xb,   // [BATCH, DIN] bf16 bits
    const unsigned short* __restrict__ wb,   // [DOUT, DIN] bf16 bits (K-major rows)
    const float* __restrict__ bias,          // [DOUT]
    float* __restrict__ y)                   // [BATCH, DOUT] f32
{
  __shared__ alignas(16) unsigned short As[2][BM][LDK];   // 20 KB
  __shared__ alignas(16) unsigned short Bs[2][BN][LDK];   // 40 KB

  const int tid   = threadIdx.x;
  const int lane  = tid & 31;
  const int wid   = tid >> 5;
  const int waveM = wid & 1;        // 0..1 -> 64 rows each
  const int waveN = wid >> 1;       // 0..3 -> 64 cols each
  const int bM = blockIdx.y * BM;
  const int bN = blockIdx.x * BN;

  // global->LDS: 4 threads per 64B row chunk, 64 rows per pass
  const int lrow = tid >> 2;        // 0..63
  const int lcol = (tid & 3) * 8;   // element offset {0,8,16,24}

  const int fr    = lane & 15;      // fragment row within 16
  const int hi16  = lane >> 4;      // 0 for lanes 0-15, 1 for 16-31
  const int kaOff = hi16 * 8;       // A: elements {ka..ka+7} U {ka+16..ka+23}
  const int kbOff = hi16 * 8;       // B: elements {kb..kb+7} U {kb+16..kb+23}

  v8f acc[4][4] = {};

#if defined(HAVE_ASYNC_LDS)
  // ---- async copy: global -> LDS directly, tracked by ASYNCcnt ----
  #define ISSUE_TILE(ktile, buf)                                                     \
    do {                                                                             \
      _Pragma("unroll")                                                              \
      for (int p = 0; p < 2; ++p) {                                                  \
        int row = lrow + p * 64;                                                     \
        __builtin_amdgcn_global_load_async_to_lds_b128(                              \
            (AS_GLOBAL v4i*)(xb + (size_t)(bM + row) * DIN + (ktile) * BK + lcol),   \
            (AS_LDS v4i*)(&As[buf][row][lcol]), 0, 0);                               \
      }                                                                              \
      _Pragma("unroll")                                                              \
      for (int p = 0; p < 4; ++p) {                                                  \
        int row = lrow + p * 64;                                                     \
        __builtin_amdgcn_global_load_async_to_lds_b128(                              \
            (AS_GLOBAL v4i*)(wb + (size_t)(bN + row) * DIN + (ktile) * BK + lcol),   \
            (AS_LDS v4i*)(&Bs[buf][row][lcol]), 0, 0);                               \
      }                                                                              \
    } while (0)

  ISSUE_TILE(0, 0);
  wait_async0();
  __syncthreads();

  const int NK = DIN / BK;   // 128
  for (int kt = 0; kt < NK; ++kt) {
    const int cur = kt & 1, nxt = cur ^ 1;
    if (kt + 1 < NK) ISSUE_TILE(kt + 1, nxt);

    v16bf afrag[4], bfrag[4];
#pragma unroll
    for (int mi = 0; mi < 4; ++mi) {
      int r = waveM * 64 + mi * 16 + fr;
      Frag32 f;
      f.lo = *(const uint4*)(&As[cur][r][kaOff]);
      f.hi = *(const uint4*)(&As[cur][r][kaOff + 16]);
      afrag[mi] = __builtin_bit_cast(v16bf, f);
    }
#pragma unroll
    for (int ni = 0; ni < 4; ++ni) {
      int r = waveN * 64 + ni * 16 + fr;
      Frag32 f;
      f.lo = *(const uint4*)(&Bs[cur][r][kbOff]);
      f.hi = *(const uint4*)(&Bs[cur][r][kbOff + 16]);
      bfrag[ni] = __builtin_bit_cast(v16bf, f);
    }
#pragma unroll
    for (int mi = 0; mi < 4; ++mi)
#pragma unroll
      for (int ni = 0; ni < 4; ++ni)
        acc[mi][ni] = __builtin_amdgcn_wmma_f32_16x16x32_bf16(
            false, afrag[mi], false, bfrag[ni], (short)0, acc[mi][ni], false, false);

    wait_async0();
    __syncthreads();
  }
  #undef ISSUE_TILE
#else
  // ---- fallback: staged global->VGPR->LDS double buffering ----
  uint4 rA[2], rB[4];
#pragma unroll
  for (int p = 0; p < 2; ++p)
    rA[p] = *(const uint4*)(xb + (size_t)(bM + lrow + p * 64) * DIN + lcol);
#pragma unroll
  for (int p = 0; p < 4; ++p)
    rB[p] = *(const uint4*)(wb + (size_t)(bN + lrow + p * 64) * DIN + lcol);
#pragma unroll
  for (int p = 0; p < 2; ++p) *(uint4*)(&As[0][lrow + p * 64][lcol]) = rA[p];
#pragma unroll
  for (int p = 0; p < 4; ++p) *(uint4*)(&Bs[0][lrow + p * 64][lcol]) = rB[p];
  __syncthreads();

  const int NK = DIN / BK;   // 128
  for (int kt = 0; kt < NK; ++kt) {
    const int cur = kt & 1, nxt = cur ^ 1;
    if (kt + 1 < NK) {
#pragma unroll
      for (int p = 0; p < 2; ++p) {
        const unsigned short* pa =
            xb + (size_t)(bM + lrow + p * 64) * DIN + (kt + 1) * BK + lcol;
        rA[p] = *(const uint4*)pa;
        if (kt + 2 < NK) __builtin_prefetch(pa + BK, 0, 3);
      }
#pragma unroll
      for (int p = 0; p < 4; ++p) {
        const unsigned short* pb =
            wb + (size_t)(bN + lrow + p * 64) * DIN + (kt + 1) * BK + lcol;
        rB[p] = *(const uint4*)pb;
        if (kt + 2 < NK) __builtin_prefetch(pb + BK, 0, 3);
      }
    }

    v16bf afrag[4], bfrag[4];
#pragma unroll
    for (int mi = 0; mi < 4; ++mi) {
      int r = waveM * 64 + mi * 16 + fr;
      Frag32 f;
      f.lo = *(const uint4*)(&As[cur][r][kaOff]);
      f.hi = *(const uint4*)(&As[cur][r][kaOff + 16]);
      afrag[mi] = __builtin_bit_cast(v16bf, f);
    }
#pragma unroll
    for (int ni = 0; ni < 4; ++ni) {
      int r = waveN * 64 + ni * 16 + fr;
      Frag32 f;
      f.lo = *(const uint4*)(&Bs[cur][r][kbOff]);
      f.hi = *(const uint4*)(&Bs[cur][r][kbOff + 16]);
      bfrag[ni] = __builtin_bit_cast(v16bf, f);
    }
#pragma unroll
    for (int mi = 0; mi < 4; ++mi)
#pragma unroll
      for (int ni = 0; ni < 4; ++ni)
        acc[mi][ni] = __builtin_amdgcn_wmma_f32_16x16x32_bf16(
            false, afrag[mi], false, bfrag[ni], (short)0, acc[mi][ni], false, false);

    if (kt + 1 < NK) {
#pragma unroll
      for (int p = 0; p < 2; ++p) *(uint4*)(&As[nxt][lrow + p * 64][lcol]) = rA[p];
#pragma unroll
      for (int p = 0; p < 4; ++p) *(uint4*)(&Bs[nxt][lrow + p * 64][lcol]) = rB[p];
    }
    __syncthreads();
  }
#endif

  // epilogue: C layout (f32 16x16): VGPR v -> row v (+8 for lanes>=16), col = lane%16
#pragma unroll
  for (int ni = 0; ni < 4; ++ni) {
    int col = bN + waveN * 64 + ni * 16 + fr;
    float bv = bias[col];
#pragma unroll
    for (int mi = 0; mi < 4; ++mi) {
      int rbase = bM + waveM * 64 + mi * 16 + hi16 * 8;
#pragma unroll
      for (int v = 0; v < 8; ++v)
        y[(size_t)(rbase + v) * DOUT + col] = acc[mi][ni][v] + bv;
    }
  }
}

// ---------------- batchnorm stats: per-column partial sums ----------------
__global__ void bn_stats_partial(const float* __restrict__ y,
                                 float* __restrict__ psum, float* __restrict__ psumsq) {
  int col = blockIdx.x * 256 + threadIdx.x;
  int r0  = blockIdx.y * 256;
  float s = 0.f, ss = 0.f;
  for (int r = r0; r < r0 + 256; ++r) {
    float v = y[(size_t)r * DOUT + col];
    s += v; ss += v * v;
  }
  psum[(size_t)blockIdx.y * DOUT + col]   = s;
  psumsq[(size_t)blockIdx.y * DOUT + col] = ss;
}

__global__ void bn_finalize(const float* __restrict__ psum, const float* __restrict__ psumsq,
                            float* __restrict__ mean, float* __restrict__ rstd) {
  int col = blockIdx.x * 256 + threadIdx.x;
  float s = 0.f, ss = 0.f;
#pragma unroll
  for (int p = 0; p < 32; ++p) {
    s  += psum[(size_t)p * DOUT + col];
    ss += psumsq[(size_t)p * DOUT + col];
  }
  float m   = s * (1.f / BATCH);
  float var = fmaxf(ss * (1.f / BATCH) - m * m, 0.f);
  mean[col] = m;
  rstd[col] = rsqrtf(var + 1e-5f);
}

// ---------------- out = sign((y-mean)*rstd*gamma + beta) ----------------
__global__ void bn_sign_out(const float* __restrict__ y, const float* __restrict__ mean,
                            const float* __restrict__ rstd, const float* __restrict__ gamma,
                            const float* __restrict__ beta, float* __restrict__ out) {
  size_t i = ((size_t)blockIdx.x * blockDim.x + threadIdx.x) * 4;
  int col = (int)(i & (size_t)(DOUT - 1));
  float4 v  = *(const float4*)(y + i);
  float4 m  = *(const float4*)(mean + col);
  float4 r  = *(const float4*)(rstd + col);
  float4 g  = *(const float4*)(gamma + col);
  float4 be = *(const float4*)(beta + col);
  float4 o;
  o.x = sgnf((v.x - m.x) * r.x * g.x + be.x);
  o.y = sgnf((v.y - m.y) * r.y * g.y + be.y);
  o.z = sgnf((v.z - m.z) * r.z * g.z + be.z);
  o.w = sgnf((v.w - m.w) * r.w * g.w + be.w);
  *(float4*)(out + i) = o;
}

extern "C" void kernel_launch(void* const* d_in, const int* in_sizes, int n_in,
                              void* d_out, int out_size, void* d_ws, size_t ws_size,
                              hipStream_t stream) {
  (void)in_sizes; (void)n_in; (void)out_size; (void)ws_size;
  const float* x     = (const float*)d_in[0];
  const float* W     = (const float*)d_in[1];
  const float* b     = (const float*)d_in[2];
  const float* gamma = (const float*)d_in[3];
  const float* beta  = (const float*)d_in[4];
  float* out = (float*)d_out;

  char* ws = (char*)d_ws;
  unsigned short* xb = (unsigned short*)ws;                                        // 64 MB
  unsigned short* wb = (unsigned short*)(ws + (size_t)BATCH * DIN * 2);            // 32 MB
  float* y    = (float*)(ws + (size_t)BATCH * DIN * 2 + (size_t)DOUT * DIN * 2);   // 128 MB
  float* ps   = (float*)((char*)y + (size_t)BATCH * DOUT * 4);
  float* pss  = ps  + (size_t)32 * DOUT;
  float* mean = pss + (size_t)32 * DOUT;
  float* rstd = mean + DOUT;

  cvt_x_bf16<<<(BATCH * DIN) / (256 * 8), 256, 0, stream>>>(x, xb);
  cvt_w_sign_bf16<<<(DOUT * DIN) / (256 * 8), 256, 0, stream>>>(W, wb);
  bingemm<<<dim3(DOUT / BN, BATCH / BM), 256, 0, stream>>>(xb, wb, b, y);
  bn_stats_partial<<<dim3(DOUT / 256, 32), 256, 0, stream>>>(y, ps, pss);
  bn_finalize<<<DOUT / 256, 256, 0, stream>>>(ps, pss, mean, rstd);
  bn_sign_out<<<(BATCH * DOUT) / (256 * 4), 256, 0, stream>>>(y, mean, rstd, gamma, beta, out);
}
